// _UnimolDescriminatorLayer_38070590112237
// MI455X (gfx1250) — matrix operand order repaired and checked by reading, not compiled, not run on hardware
//
#include <hip/hip_runtime.h>
#include <hip/hip_fp16.h>

typedef __attribute__((ext_vector_type(16))) _Float16 v16h;
typedef __attribute__((ext_vector_type(8)))  _Float16 v8h;
typedef __attribute__((ext_vector_type(8)))  float    v8f;

#define L_SEQ 512
#define BATCH 8
#define DIM   512
#define NHEAD 16
#define HDIM  32
#define BHEAD (BATCH*NHEAD)   // 128
#define MROWS (L_SEQ*BATCH)   // 4096

// ---------------------------------------------------------------------------
// WMMA operand loaders (wave32, per CDNA5 ISA §7.12.2)
// ---------------------------------------------------------------------------

// 16-bit A matrix 16x32 (MxK), row-major source A[row][k], leading dim lda.
// lanes 0-15: M=lane,   elems 0..15 -> K=k0+0..15
// lanes16-31: M=lane-16,elems 0..15 -> K=k0+16..31
__device__ __forceinline__ v16h load_a16x32(const _Float16* __restrict__ A,
                                            int lda, int row0, int k0) {
  const int lane = threadIdx.x & 31;
  const int hf = lane >> 4;
  const int m  = lane & 15;
  const _Float16* base = A + (size_t)(row0 + m) * lda + k0;
  v8h lo = *reinterpret_cast<const v8h*>(base + 16 * hf);
  v8h hi = *reinterpret_cast<const v8h*>(base + 16 * hf + 8);
  v16h a;
#pragma unroll
  for (int i = 0; i < 8; ++i) { a[i] = lo[i]; a[8 + i] = hi[i]; }
  return a;
}

// 16-bit B matrix 32x16 (KxN) where B = W^T, W row-major [N][K] (ldw).
// lanes 0-15 hold K=k0+0..15 in elems 0..15; lanes 16-31 hold K=k0+16..31.
// N = lane & 15 -> B[k][n] = W[n0+n][k] : one contiguous 32B load per lane.
__device__ __forceinline__ v16h load_bT32x16(const _Float16* __restrict__ W,
                                             int ldw, int n0, int k0) {
  const int lane = threadIdx.x & 31;
  const int hf = lane >> 4;
  const int n  = lane & 15;
  const _Float16* base = W + (size_t)(n0 + n) * ldw + k0 + 16 * hf;
  return *reinterpret_cast<const v16h*>(base);
}

__device__ __forceinline__ v8f wmma_f16(v16h a, v16h b, v8f c) {
  // (neg_a, A, neg_b, B, c_mod, C, reuse_a, reuse_b)
  return __builtin_amdgcn_wmma_f32_16x16x32_f16(false, a, false, b, (short)0, c,
                                                false, false);
}

__device__ __forceinline__ float gelu_exact(float x) {
  return 0.5f * x * (1.0f + erff(x * 0.70710678118654752f));
}

// ---------------------------------------------------------------------------
// Elementwise / reduction kernels
// ---------------------------------------------------------------------------

__global__ void k_f32_to_f16(const float* __restrict__ x, _Float16* __restrict__ y, int n) {
  int i = blockIdx.x * 256 + threadIdx.x;
  if (i < n) y[i] = (_Float16)x[i];
}

// LayerNorm over last dim (512). One block (256 thr) per row. Optional f32/f16 outs.
__global__ void k_layernorm(const float* __restrict__ x, const float* __restrict__ g,
                            const float* __restrict__ b, float* __restrict__ y32,
                            _Float16* __restrict__ y16) {
  __shared__ float s1[256], s2[256];
  const int row = blockIdx.x, t = threadIdx.x;
  const float* xr = x + (size_t)row * DIM;
  float v0 = xr[t], v1 = xr[t + 256];
  s1[t] = v0 + v1;
  s2[t] = v0 * v0 + v1 * v1;
  __syncthreads();
  for (int off = 128; off > 0; off >>= 1) {
    if (t < off) { s1[t] += s1[t + off]; s2[t] += s2[t + off]; }
    __syncthreads();
  }
  float mean = s1[0] * (1.0f / DIM);
  float var  = s2[0] * (1.0f / DIM) - mean * mean;
  float rstd = rsqrtf(var + 1e-5f);
  float o0 = (v0 - mean) * rstd * g[t]       + b[t];
  float o1 = (v1 - mean) * rstd * g[t + 256] + b[t + 256];
  size_t o = (size_t)row * DIM;
  if (y32) { y32[o + t] = o0; y32[o + t + 256] = o1; }
  if (y16) { y16[o + t] = (_Float16)o0; y16[o + t + 256] = (_Float16)o1; }
}

// Row softmax over 512 cols of (S + E), output f16 probabilities.
__global__ void k_softmax_bias(const float* __restrict__ S, const float* __restrict__ E,
                               _Float16* __restrict__ P) {
  __shared__ float red[256];
  const int row = blockIdx.x, t = threadIdx.x;
  const float* sr = S + (size_t)row * L_SEQ;
  const float* er = E + (size_t)row * L_SEQ;
  float a0 = sr[t] + er[t];
  float a1 = sr[t + 256] + er[t + 256];
  red[t] = fmaxf(a0, a1);
  __syncthreads();
  for (int off = 128; off > 0; off >>= 1) {
    if (t < off) red[t] = fmaxf(red[t], red[t + off]);
    __syncthreads();
  }
  float mx = red[0];
  __syncthreads();
  float e0 = __expf(a0 - mx), e1 = __expf(a1 - mx);
  red[t] = e0 + e1;
  __syncthreads();
  for (int off = 128; off > 0; off >>= 1) {
    if (t < off) red[t] += red[t + off];
    __syncthreads();
  }
  float inv = 1.0f / red[0];
  size_t o = (size_t)row * L_SEQ;
  P[o + t]       = (_Float16)(e0 * inv);
  P[o + t + 256] = (_Float16)(e1 * inv);
}

// qkv [4096,1536] f16 -> q[bh][l][hd]*scale, k[bh][l][hd], vT[bh][hd][l]
__global__ void k_split_qkv(const _Float16* __restrict__ qkv, _Float16* __restrict__ q,
                            _Float16* __restrict__ k, _Float16* __restrict__ vt) {
  const float scale = 0.17677669529663687f; // 1/sqrt(HDIM)
  int i = blockIdx.x * 256 + threadIdx.x;   // < 4096*1536
  int m = i / 1536, c = i - m * 1536;
  int which = c / DIM, d = c - which * DIM;
  int h = d >> 5, hd = d & 31;
  int l = m >> 3, bb = m & 7;
  int bh = bb * NHEAD + h;
  _Float16 v = qkv[i];
  if (which == 0)      q[((size_t)bh * L_SEQ + l) * HDIM + hd] = (_Float16)((float)v * scale);
  else if (which == 1) k[((size_t)bh * L_SEQ + l) * HDIM + hd] = v;
  else                 vt[((size_t)bh * HDIM + hd) * L_SEQ + l] = v;
}

// x [4096,512] f16 -> out[bh][l][hd] * scale
__global__ void k_split_heads(const _Float16* __restrict__ x, _Float16* __restrict__ out,
                              float scale) {
  int i = blockIdx.x * 256 + threadIdx.x;   // < 4096*512
  int m = i >> 9, d = i & 511;
  int h = d >> 5, hd = d & 31;
  int l = m >> 3, bb = m & 7;
  int bh = bb * NHEAD + h;
  out[((size_t)bh * L_SEQ + l) * HDIM + hd] = (_Float16)((float)x[i] * scale);
}

// ---------------------------------------------------------------------------
// WMMA GEMM:  C[M,N] = act( A[M,K](f16) @ W[N,K]^T(f16) + bias + resid )
// block = 256 thr = 8 waves; each wave owns a 32x64 strip:
//   2 M-tiles x 4 N-tiles -> 8 WMMAs per K-step vs 4 A-loads + 8 B-loads (b128).
// Next K-slab of streamed A rows is prefetched (global_prefetch_b8); W is small
// enough (<= 2 MB f16 per layer) to stay resident in the 192 MB L2.
// ---------------------------------------------------------------------------
template <bool OUT16, bool GELU_ACT, bool RESID>
__global__ void k_gemm_wmma(const _Float16* __restrict__ A, const _Float16* __restrict__ W,
                            const float* __restrict__ bias, const float* __restrict__ resid,
                            float* __restrict__ C32, _Float16* __restrict__ C16,
                            int M, int N, int K) {
  const int wave = blockIdx.x * 8 + (threadIdx.x >> 5);
  const int strips = N >> 6;               // N / 64
  const int tm = wave / strips;            // 32-row block index
  const int n0 = (wave - tm * strips) << 6;
  const int row0 = tm * 32;
  if (row0 >= M) return;
  const int lane = threadIdx.x & 31;
  const int hf = lane >> 4, ln = lane & 15;

  v8f acc[2][4];
#pragma unroll
  for (int j = 0; j < 4; ++j) {
    float bv = bias ? bias[n0 + j * 16 + ln] : 0.0f;
#pragma unroll
    for (int r = 0; r < 8; ++r) { acc[0][j][r] = bv; acc[1][j][r] = bv; }
  }
  for (int k0 = 0; k0 < K; k0 += 32) {
    if (k0 + 32 < K) {
      // each lane prefetches one of the 32 A rows of this strip, next K-slab
      __builtin_prefetch(A + (size_t)(row0 + lane) * K + k0 + 32, 0, 3);
    }
    v16h a0 = load_a16x32(A, K, row0, k0);
    v16h a1 = load_a16x32(A, K, row0 + 16, k0);
#pragma unroll
    for (int j = 0; j < 4; ++j) {
      v16h b = load_bT32x16(W, K, n0 + j * 16, k0);
      acc[0][j] = wmma_f16(a0, b, acc[0][j]);
      acc[1][j] = wmma_f16(a1, b, acc[1][j]);
    }
  }
#pragma unroll
  for (int mi = 0; mi < 2; ++mi) {
#pragma unroll
    for (int j = 0; j < 4; ++j) {
      const int col = n0 + j * 16 + ln;
#pragma unroll
      for (int r = 0; r < 8; ++r) {
        const int row = row0 + mi * 16 + 8 * hf + r;
        size_t idx = (size_t)row * N + col;
        float v = acc[mi][j][r];
        if (RESID)    v += resid[idx];
        if (GELU_ACT) v = gelu_exact(v);
        if (OUT16) C16[idx] = (_Float16)v;
        else       C32[idx] = v;
      }
    }
  }
}

// ---------------------------------------------------------------------------
// Batched attention scores: S[bh] = q[bh](512x32) @ k[bh](512x32)^T  (scale in q)
// one wave per 16x16 tile; single K=32 WMMA.
// ---------------------------------------------------------------------------
__global__ void k_attn_scores(const _Float16* __restrict__ Q, const _Float16* __restrict__ Km,
                              float* __restrict__ S) {
  const int wave = blockIdx.x * 8 + (threadIdx.x >> 5);
  const int bh = wave >> 10;               // 1024 tiles per batch-head
  const int t = wave & 1023;
  const int tm = t >> 5, tn = t & 31;
  const _Float16* Qb = Q  + (size_t)bh * L_SEQ * HDIM;
  const _Float16* Kb = Km + (size_t)bh * L_SEQ * HDIM;
  v16h a = load_a16x32(Qb, HDIM, tm * 16, 0);
  v16h b = load_bT32x16(Kb, HDIM, tn * 16, 0);
  v8f c = {};
  c = wmma_f16(a, b, c);
  const int lane = threadIdx.x & 31;
  const int hf = lane >> 4, ln = lane & 15;
  float* Sb = S + (size_t)bh * L_SEQ * L_SEQ;
#pragma unroll
  for (int r = 0; r < 8; ++r)
    Sb[(size_t)(tm * 16 + 8 * hf + r) * L_SEQ + tn * 16 + ln] = c[r];
}

// ---------------------------------------------------------------------------
// AV: ao[bh](512x32) = P[bh](512x512) @ V[bh](512x32); V stored transposed
// vt[bh][hd][l] so B loads are contiguous. Output scattered into [L*B, D] f16.
// ---------------------------------------------------------------------------
__global__ void k_attn_av(const _Float16* __restrict__ P, const _Float16* __restrict__ VT,
                          _Float16* __restrict__ AO) {
  const int wave = blockIdx.x * 8 + (threadIdx.x >> 5);
  const int bh = wave >> 6;                // 32 m-tiles x 2 n-tiles per batch-head
  const int t = wave & 63;
  const int tm = t >> 1, tn = t & 1;
  const _Float16* Pb = P  + (size_t)bh * L_SEQ * L_SEQ;
  const _Float16* Vb = VT + (size_t)bh * HDIM * L_SEQ;
  v8f acc = {};
  for (int k0 = 0; k0 < L_SEQ; k0 += 32) {
    if (k0 + 32 < L_SEQ)
      __builtin_prefetch(Pb + (size_t)(tm * 16 + (threadIdx.x & 15)) * L_SEQ + k0 + 32, 0, 3);
    v16h a = load_a16x32(Pb, L_SEQ, tm * 16, k0);
    v16h b = load_bT32x16(Vb, L_SEQ, tn * 16, k0);   // B[k][n] = vt[n][k]
    acc = wmma_f16(a, b, acc);
  }
  const int lane = threadIdx.x & 31;
  const int hf = lane >> 4, ln = lane & 15;
  const int bb = bh >> 4, h = bh & 15;
  const int col = h * HDIM + tn * 16 + ln;
#pragma unroll
  for (int r = 0; r < 8; ++r) {
    int l = tm * 16 + 8 * hf + r;
    AO[((size_t)l * BATCH + bb) * DIM + col] = (_Float16)acc[r];
  }
}

// ---------------------------------------------------------------------------
// Fused: edge_out = edge0 + sa_t + (q2 @ k2^T)   (mha_t never materialized)
// ---------------------------------------------------------------------------
__global__ void k_edge_fuse(const _Float16* __restrict__ Q2, const _Float16* __restrict__ K2,
                            const float* __restrict__ E0, const float* __restrict__ SA,
                            float* __restrict__ EO) {
  const int wave = blockIdx.x * 8 + (threadIdx.x >> 5);
  const int bh = wave >> 10;
  const int t = wave & 1023;
  const int tm = t >> 5, tn = t & 31;
  const _Float16* Qb = Q2 + (size_t)bh * L_SEQ * HDIM;
  const _Float16* Kb = K2 + (size_t)bh * L_SEQ * HDIM;
  v16h a = load_a16x32(Qb, HDIM, tm * 16, 0);
  v16h b = load_bT32x16(Kb, HDIM, tn * 16, 0);
  v8f c = {};
  c = wmma_f16(a, b, c);
  const int lane = threadIdx.x & 31;
  const int hf = lane >> 4, ln = lane & 15;
  size_t base = (size_t)bh * L_SEQ * L_SEQ + (size_t)(tm * 16 + 8 * hf) * L_SEQ
              + tn * 16 + ln;
#pragma unroll
  for (int r = 0; r < 8; ++r) {
    size_t idx = base + (size_t)r * L_SEQ;
    EO[idx] = E0[idx] + SA[idx] + c[r];
  }
}

// ---------------------------------------------------------------------------
// Host side
// ---------------------------------------------------------------------------
extern "C" void kernel_launch(void* const* d_in, const int* in_sizes, int n_in,
                              void* d_out, int out_size, void* d_ws, size_t ws_size,
                              hipStream_t stream) {
  (void)in_sizes; (void)n_in; (void)out_size; (void)ws_size;
  const float* node0      = (const float*)d_in[0];
  const float* edge0      = (const float*)d_in[1];
  const float* node1      = (const float*)d_in[2];
  /* d_in[3] = edge1 : unused by reference */
  const float* in_proj_w  = (const float*)d_in[4];
  const float* in_proj_b  = (const float*)d_in[5];
  const float* out_proj_w = (const float*)d_in[6];
  const float* out_proj_b = (const float*)d_in[7];
  const float* mha_q_w    = (const float*)d_in[8];
  const float* mha_q_b    = (const float*)d_in[9];
  const float* mha_kv_w   = (const float*)d_in[10];
  const float* mha_kv_b   = (const float*)d_in[11];
  const float* lin1_w     = (const float*)d_in[12];
  const float* lin1_b     = (const float*)d_in[13];
  const float* lin2_w     = (const float*)d_in[14];
  const float* lin2_b     = (const float*)d_in[15];
  const float* norm1_w    = (const float*)d_in[16];
  const float* norm1_b    = (const float*)d_in[17];
  const float* mha_norm_w = (const float*)d_in[18];
  const float* mha_norm_b = (const float*)d_in[19];
  const float* norm2_w    = (const float*)d_in[20];
  const float* norm2_b    = (const float*)d_in[21];

  float* x_out    = (float*)d_out;                       // [512,8,512]
  float* edge_out = (float*)d_out + (size_t)MROWS * DIM; // [128,512,512]

  // workspace layout (bytes, 256-aligned)
  char* ws = (char*)d_ws;
  _Float16* w_inproj_h  = (_Float16*)(ws + 0);           // 1536x512
  _Float16* w_outproj_h = (_Float16*)(ws + 1572864);     // 512x512
  _Float16* w_mhaq_h    = (_Float16*)(ws + 2097152);     // 512x512
  _Float16* w_mhak_h    = (_Float16*)(ws + 2621440);     // 512x512 (k half of kv)
  _Float16* w_lin1_h    = (_Float16*)(ws + 3145728);     // 2048x512
  _Float16* w_lin2_h    = (_Float16*)(ws + 5242880);     // 512x2048
  _Float16* ln0_h       = (_Float16*)(ws + 7340032);     // 4096x512
  _Float16* qkv_h       = (_Float16*)(ws + 11534336);    // 4096x1536
  _Float16* q_h         = (_Float16*)(ws + 24117248);    // [128][512][32]
  _Float16* k_h         = (_Float16*)(ws + 28311552);
  _Float16* vt_h        = (_Float16*)(ws + 32505856);    // [128][32][512]
  _Float16* ao_h        = (_Float16*)(ws + 36700160);    // 4096x512
  float*    x_f32       = (float*)   (ws + 40894464);    // 4096x512
  float*    xn_f32      = (float*)   (ws + 49283072);
  _Float16* xn_h        = (_Float16*)(ws + 57671680);
  _Float16* tmp_h       = (_Float16*)(ws + 61865984);    // 4096x512 staging
  _Float16* q2_h        = (_Float16*)(ws + 66060288);
  _Float16* k2_h        = (_Float16*)(ws + 70254592);
  _Float16* node1_h     = (_Float16*)(ws + 74448896);
  _Float16* ln2_h       = (_Float16*)(ws + 78643200);
  _Float16* hff_h       = (_Float16*)(ws + 82837504);    // 4096x2048
  _Float16* aw_h        = (_Float16*)(ws + 99614720);    // [128][512][512]
  float*    sa_f32      = (float*)   (ws + 166723584);   // [128][512][512]

  const int B256 = 256;
  auto cvt = [&](const float* src, _Float16* dst, int n) {
    k_f32_to_f16<<<(n + 255) / 256, B256, 0, stream>>>(src, dst, n);
  };

  // --- weight & input conversions to f16 ---
  cvt(in_proj_w,  w_inproj_h,  1536 * 512);
  cvt(out_proj_w, w_outproj_h, 512 * 512);
  cvt(mha_q_w,    w_mhaq_h,    512 * 512);
  cvt(mha_kv_w,   w_mhak_h,    512 * 512);      // rows 0..511 = k-projection
  cvt(lin1_w,     w_lin1_h,    2048 * 512);
  cvt(lin2_w,     w_lin2_h,    512 * 2048);
  cvt(node1,      node1_h,     MROWS * DIM);

  // --- self-attention branch ---
  k_layernorm<<<MROWS, B256, 0, stream>>>(node0, norm1_w, norm1_b, nullptr, ln0_h);
  // qkv = ln0 @ in_proj^T + b : M=4096 N=1536 K=512 -> (128*24)=3072 waves -> 384 blocks
  k_gemm_wmma<true, false, false><<<384, B256, 0, stream>>>(
      ln0_h, w_inproj_h, in_proj_b, nullptr, nullptr, qkv_h, MROWS, 1536, 512);
  k_split_qkv<<<(MROWS * 1536) / 256, B256, 0, stream>>>(qkv_h, q_h, k_h, vt_h);
  // sa_t = q @ k^T (scale folded into q): 128*1024 tiles / 8 waves
  k_attn_scores<<<16384, B256, 0, stream>>>(q_h, k_h, sa_f32);
  k_softmax_bias<<<BHEAD * L_SEQ, B256, 0, stream>>>(sa_f32, edge0, aw_h);
  k_attn_av<<<1024, B256, 0, stream>>>(aw_h, vt_h, ao_h);
  // x = node0 + ao @ out_proj^T + b : (128*8)=1024 waves -> 128 blocks
  k_gemm_wmma<false, false, true><<<128, B256, 0, stream>>>(
      ao_h, w_outproj_h, out_proj_b, node0, x_f32, nullptr, MROWS, DIM, 512);

  // --- cross-attention scores (only raw scores used) ---
  k_layernorm<<<MROWS, B256, 0, stream>>>(x_f32, mha_norm_w, mha_norm_b, xn_f32, xn_h);
  k_gemm_wmma<true, false, false><<<128, B256, 0, stream>>>(
      xn_h, w_mhaq_h, mha_q_b, nullptr, nullptr, tmp_h, MROWS, DIM, 512);
  k_split_heads<<<(MROWS * DIM) / 256, B256, 0, stream>>>(tmp_h, q2_h,
                                                          0.17677669529663687f);
  k_gemm_wmma<true, false, false><<<128, B256, 0, stream>>>(
      node1_h, w_mhak_h, mha_kv_b, nullptr, nullptr, tmp_h, MROWS, DIM, 512);
  k_split_heads<<<(MROWS * DIM) / 256, B256, 0, stream>>>(tmp_h, k2_h, 1.0f);
  // edge_out = edge0 + sa_t + q2@k2^T  (fused, mha_t never stored)
  k_edge_fuse<<<16384, B256, 0, stream>>>(q2_h, k2_h, edge0, sa_f32, edge_out);

  // --- FFN (residual base = xn) ---
  k_layernorm<<<MROWS, B256, 0, stream>>>(xn_f32, norm2_w, norm2_b, nullptr, ln2_h);
  // h = gelu(ln2 @ lin1^T + b): (128*32)=4096 waves -> 512 blocks
  k_gemm_wmma<true, true, false><<<512, B256, 0, stream>>>(
      ln2_h, w_lin1_h, lin1_b, nullptr, nullptr, hff_h, MROWS, 2048, 512);
  // x_out = xn + h @ lin2^T + b: N=512 K=2048 -> 128 blocks
  k_gemm_wmma<false, false, true><<<128, B256, 0, stream>>>(
      hff_h, w_lin2_h, lin2_b, xn_f32, x_out, nullptr, MROWS, DIM, 2048);
}
